// VRNN_21268678050172
// MI455X (gfx1250) — matrix operand order repaired and checked by reading, not compiled, not run on hardware
//
#include <hip/hip_runtime.h>
#include <hip/hip_bf16.h>
#include <math.h>

// ---------------- problem constants ----------------
#define TSTEPS 16384
#define INDIM  1024
#define LDIM   512
#define LOG2PI 1.8378770664093453f

// ---------------- recurrence sizing ----------------
#define NB  128        // persistent blocks (co-resident: tiny LDS/VGPR footprint)
#define TPB 256        // 8 wave32 per block -> 1024 waves total

typedef __attribute__((ext_vector_type(16))) __bf16 v16bf;
typedef __attribute__((ext_vector_type(8)))  float  v8f;

// ================= scratch layout (floats) =================
// [0 .. T*512)                      : xhat_all
// then: h(512) c(512) pr(1024) enc(1024) zhat(512) gates(2048) wp(1024) bar(2 uint)
#define WS_XHAT   0
#define WS_BASE   ((size_t)TSTEPS * LDIM)
#define WS_H      (WS_BASE)
#define WS_C      (WS_H + LDIM)
#define WS_PR     (WS_C + LDIM)
#define WS_ENC    (WS_PR + 2*LDIM)
#define WS_ZH     (WS_ENC + 2*LDIM)
#define WS_GATES  (WS_ZH + LDIM)
#define WS_WP     (WS_GATES + 4*LDIM)
#define WS_BAR    (WS_WP + 2*LDIM)

// ================= init: zero h, c, barrier =================
__global__ __launch_bounds__(256) void vrnn_init(float* ws) {
    int tid = threadIdx.x;
    for (int i = tid; i < 2*LDIM; i += 256) ws[WS_H + i] = 0.0f;   // h then c
    if (tid < 2) ((unsigned*)(ws + WS_BAR))[tid] = 0u;
}

// ================= phase 1: x_hat GEMM via WMMA bf16x3 =================
// x_hat[t,n] = relu( sum_k X[t,k]*W[n,k] + b[n] ),  M=16384 N=512 K=1024
// One 16x16 output tile per wave; K in chunks of 32; bf16 hi/lo split of both
// operands -> 3x v_wmma_f32_16x16x32_bf16 per chunk, f32 accumulate.
__global__ __launch_bounds__(256) void xhat_gemm(const float* __restrict__ X,
                                                 const float* __restrict__ W,
                                                 const float* __restrict__ b,
                                                 float* __restrict__ xhat) {
    const int lane   = threadIdx.x & 31;
    const int gw     = blockIdx.x * 8 + (threadIdx.x >> 5);
    const int tile_n = gw & 31;                 // N=512 -> 32 tiles
    const int tile_m = gw >> 5;                 // M=16384 -> 1024 tiles
    const int m      = tile_m * 16 + (lane & 15);
    const int n      = tile_n * 16 + (lane & 15);
    const int koffA  = (lane >> 4) * 8;         // A frag: K runs {0..7,16..23} / {8..15,24..31}
    const int koffB  = (lane >> 4) * 8;         // B frag: K runs {0..7,16..23} / {8..15,24..31}

    const float* __restrict__ arow = X + (size_t)m * INDIM;
    const float* __restrict__ brow = W + (size_t)n * INDIM;   // B[k,n] = W[n,k]

    v8f acc = {};
    for (int kb = 0; kb < INDIM; kb += 32) {
        // prefetch A chunk two ahead (global_prefetch_b8)
        if (kb + 64 < INDIM) __builtin_prefetch(arow + kb + 64, 0, 1);

        float av[16], bv[16];
#pragma unroll
        for (int j = 0; j < 8; ++j) {
            av[j]     = arow[kb + koffA + j];
            av[8 + j] = arow[kb + 16 + koffA + j];
        }
#pragma unroll
        for (int j = 0; j < 8; ++j) { bv[j] = brow[kb + koffB + j]; bv[8 + j] = brow[kb + 16 + koffB + j]; }

        v16bf ah, al, bh, bl;
#pragma unroll
        for (int j = 0; j < 16; ++j) {
            float a = av[j];  __bf16 h  = (__bf16)a;  ah[j] = h;  al[j] = (__bf16)(a - (float)h);
            float w = bv[j];  __bf16 hw = (__bf16)w;  bh[j] = hw; bl[j] = (__bf16)(w - (float)hw);
        }
        acc = __builtin_amdgcn_wmma_f32_16x16x32_bf16(false, ah, false, bh, (short)0, acc, false, false);
        acc = __builtin_amdgcn_wmma_f32_16x16x32_bf16(false, ah, false, bl, (short)0, acc, false, false);
        acc = __builtin_amdgcn_wmma_f32_16x16x32_bf16(false, al, false, bh, (short)0, acc, false, false);
    }
    // C/D layout: VGPR r -> lanes0-15 (M=r), lanes16-31 (M=8+r); N = lane&15
    const int mrow0 = tile_m * 16 + ((lane >> 4) ? 8 : 0);
    const float bias = b[n];
#pragma unroll
    for (int r = 0; r < 8; ++r) {
        float v = acc[r] + bias;
        xhat[(size_t)(mrow0 + r) * LDIM + n] = fmaxf(v, 0.0f);
    }
}

// ================= persistent recurrence =================
__device__ __forceinline__ float wave_red(float v) {
#pragma unroll
    for (int o = 16; o > 0; o >>= 1) v += __shfl_xor(v, o, 32);
    return v;
}
__device__ __forceinline__ float sigm(float x) { return 1.0f / (1.0f + expf(-x)); }

__device__ __forceinline__ void gbar(unsigned* cnt, unsigned* gen) {
    __syncthreads();
    if (threadIdx.x == 0) {
        __threadfence();
        unsigned g = __hip_atomic_load(gen, __ATOMIC_RELAXED, __HIP_MEMORY_SCOPE_AGENT);
        if (__hip_atomic_fetch_add(cnt, 1u, __ATOMIC_ACQ_REL, __HIP_MEMORY_SCOPE_AGENT) == NB - 1u) {
            __hip_atomic_store(cnt, 0u, __ATOMIC_RELAXED, __HIP_MEMORY_SCOPE_AGENT);
            __hip_atomic_store(gen, g + 1u, __ATOMIC_RELEASE, __HIP_MEMORY_SCOPE_AGENT);
        } else {
            while (__hip_atomic_load(gen, __ATOMIC_ACQUIRE, __HIP_MEMORY_SCOPE_AGENT) == g)
                __builtin_amdgcn_s_sleep(1);
        }
        __threadfence();
    }
    __syncthreads();
}

__global__ __launch_bounds__(TPB) void vrnn_recur(
    const float* __restrict__ inputs, const float* __restrict__ eps,
    const float* __restrict__ W_phiz, const float* __restrict__ b_phiz,
    const float* __restrict__ W_prior, const float* __restrict__ b_prior,
    const float* __restrict__ W_enc,  const float* __restrict__ b_enc,
    const float* __restrict__ W_dec,  const float* __restrict__ b_dec,
    const float* __restrict__ W_ih,   const float* __restrict__ W_hh,
    const float* __restrict__ b_ih,   const float* __restrict__ b_hh,
    float* __restrict__ ws, float* __restrict__ out) {

    const int tid  = threadIdx.x;
    const int lane = tid & 31;
    const int widx = tid >> 5;
    const int gw   = blockIdx.x * 8 + widx;    // 0..1023

    float* g_h     = ws + WS_H;
    float* g_c     = ws + WS_C;
    float* g_pr    = ws + WS_PR;
    float* g_enc   = ws + WS_ENC;
    float* g_zh    = ws + WS_ZH;
    float* g_gates = ws + WS_GATES;
    float* g_wp    = ws + WS_WP;
    unsigned* cnt  = (unsigned*)(ws + WS_BAR);
    unsigned* gen  = cnt + 1;
    const float* xhat = ws + WS_XHAT;

    __shared__ float s_h[LDIM], s_xh[LDIM], s_z[LDIM], s_zh[LDIM];
    __shared__ float s_r0[8], s_r1[8], s_r2[8];

    float acc_loss = 0.0f;

    for (int t = 0; t < TSTEPS; ++t) {
        const float* xh_t  = xhat + (size_t)t * LDIM;
        const float* eps_t = eps  + (size_t)t * LDIM;

        for (int i = tid; i < LDIM; i += TPB) { s_h[i] = g_h[i]; s_xh[i] = xh_t[i]; }
        __syncthreads();

        // ---- Stage A: pr (1024 rows, len 512) + enc (1024 rows, len 1024) ----
        {
            const float* wr = W_prior + (size_t)gw * LDIM;
            float d = 0.0f;
            for (int k = lane; k < LDIM; k += 32) d += wr[k] * s_h[k];
            d = wave_red(d);
            if (!lane) g_pr[gw] = fmaxf(d + b_prior[gw], 0.0f);

            const float* we = W_enc + (size_t)gw * (2*LDIM);
            float e = 0.0f;
            for (int k = lane; k < LDIM; k += 32) e += we[k] * s_h[k] + we[LDIM + k] * s_xh[k];
            e = wave_red(e);
            if (!lane) g_enc[gw] = fmaxf(e + b_enc[gw], 0.0f);
        }
        gbar(cnt, gen);

        // ---- Stage B: z (local, redundant per block) then z_hat (512 rows) ----
        for (int i = tid; i < LDIM; i += TPB) {
            float mu = g_enc[i], ls = g_enc[LDIM + i];
            s_z[i] = mu + expf(ls) * eps_t[i];
        }
        __syncthreads();
        if (gw < LDIM) {
            const float* wz = W_phiz + (size_t)gw * LDIM;
            float d = 0.0f;
            for (int k = lane; k < LDIM; k += 32) d += wz[k] * s_z[k];
            d = wave_red(d);
            if (!lane) g_zh[gw] = fmaxf(d + b_phiz[gw], 0.0f);
        }
        gbar(cnt, gen);

        // ---- Stage C: logits (1024 rows) + gates (2048 rows) ----
        for (int i = tid; i < LDIM; i += TPB) s_zh[i] = g_zh[i];
        __syncthreads();
        {
            const float* wd = W_dec + (size_t)gw * (2*LDIM);
            float d = 0.0f;
            for (int k = lane; k < LDIM; k += 32) d += wd[k] * s_zh[k] + wd[LDIM + k] * s_h[k];
            d = wave_red(d);
            if (!lane) {
                float lg = fmaxf(d + b_dec[gw], 0.0f);
                float xv = inputs[(size_t)t * INDIM + gw];
                float sp = log1pf(expf(-fabsf(lg))) + fmaxf(lg, 0.0f);   // stable softplus
                g_wp[gw] = xv * lg - sp;                                  // log_px contribution
            }
#pragma unroll
            for (int rr = 0; rr < 2; ++rr) {
                int r = gw + rr * 1024;
                const float* wi = W_ih + (size_t)r * (2*LDIM);
                const float* wh = W_hh + (size_t)r * LDIM;
                float a = 0.0f;
                for (int k = lane; k < LDIM; k += 32)
                    a += wi[k] * s_xh[k] + wi[LDIM + k] * s_zh[k] + wh[k] * s_h[k];
                a = wave_red(a);
                if (!lane) g_gates[r] = a + b_ih[r] + b_hh[r];
            }
        }
        gbar(cnt, gen);

        // ---- Stage D: LSTM update (blocks 1,2) + loss reductions (block 0) ----
        if (blockIdx.x == 1 || blockIdx.x == 2) {
            int i = (blockIdx.x - 1) * TPB + tid;            // 0..511
            float ig = g_gates[i],            fg = g_gates[LDIM + i];
            float gg = g_gates[2*LDIM + i],   og = g_gates[3*LDIM + i];
            float c2 = sigm(fg) * g_c[i] + sigm(ig) * tanhf(gg);
            float h2 = sigm(og) * tanhf(c2);
            g_c[i] = c2; g_h[i] = h2;
        } else if (blockIdx.x == 0) {
            float px = g_wp[tid] + g_wp[tid + 256] + g_wp[tid + 512] + g_wp[tid + 768];
            float qz = 0.0f, pz = 0.0f;
#pragma unroll
            for (int rr = 0; rr < 2; ++rr) {
                int i = tid + rr * 256;
                float ls_q = g_enc[LDIM + i];
                float ev   = eps_t[i];
                qz += -ls_q - 0.5f * LOG2PI - 0.5f * ev * ev;     // (z-mu_q)/sigma_q == eps
                float mu_p = g_pr[i], ls_p = g_pr[LDIM + i];
                float dd = (s_z[i] - mu_p) * expf(-ls_p);
                pz += -ls_p - 0.5f * LOG2PI - 0.5f * dd * dd;
            }
            px = wave_red(px); qz = wave_red(qz); pz = wave_red(pz);
            if (!lane) { s_r0[widx] = px; s_r1[widx] = qz; s_r2[widx] = pz; }
            __syncthreads();
            if (tid == 0) {
                float LPX = 0, LQZ = 0, LPZ = 0;
                for (int w = 0; w < 8; ++w) { LPX += s_r0[w]; LQZ += s_r1[w]; LPZ += s_r2[w]; }
                float kl   = LQZ - LPZ;
                float loss = -(LPX - kl);
                out[1 + t]              = loss;
                out[1 + TSTEPS + t]     = kl;
                out[1 + 2*TSTEPS + t]   = LPX;
                acc_loss += loss;
            }
        }
        gbar(cnt, gen);
    }
    if (blockIdx.x == 0 && tid == 0) out[0] = acc_loss;
}

// ================= launcher =================
extern "C" void kernel_launch(void* const* d_in, const int* in_sizes, int n_in,
                              void* d_out, int out_size, void* d_ws, size_t ws_size,
                              hipStream_t stream) {
    (void)in_sizes; (void)n_in; (void)out_size; (void)ws_size;
    const float* inputs  = (const float*)d_in[0];
    const float* eps     = (const float*)d_in[1];
    const float* W_phix  = (const float*)d_in[2];
    const float* b_phix  = (const float*)d_in[3];
    const float* W_phiz  = (const float*)d_in[4];
    const float* b_phiz  = (const float*)d_in[5];
    const float* W_prior = (const float*)d_in[6];
    const float* b_prior = (const float*)d_in[7];
    const float* W_enc   = (const float*)d_in[8];
    const float* b_enc   = (const float*)d_in[9];
    const float* W_dec   = (const float*)d_in[10];
    const float* b_dec   = (const float*)d_in[11];
    const float* W_ih    = (const float*)d_in[12];
    const float* W_hh    = (const float*)d_in[13];
    const float* b_ih    = (const float*)d_in[14];
    const float* b_hh    = (const float*)d_in[15];
    float* ws  = (float*)d_ws;
    float* out = (float*)d_out;

    vrnn_init<<<1, 256, 0, stream>>>(ws);

    // x_hat GEMM: 1024 x 32 = 32768 tiles, 8 waves/block -> 4096 blocks
    xhat_gemm<<<4096, 256, 0, stream>>>(inputs, W_phix, b_phix, ws + WS_XHAT);

    vrnn_recur<<<NB, TPB, 0, stream>>>(inputs, eps, W_phiz, b_phiz, W_prior, b_prior,
                                       W_enc, b_enc, W_dec, b_dec, W_ih, W_hh,
                                       b_ih, b_hh, ws, out);
}
